// EMAVectorQuantizer_41549513621530
// MI455X (gfx1250) — matrix operand order repaired and checked by reading, not compiled, not run on hardware
//
#include <hip/hip_runtime.h>
#include <stdint.h>

#define N_TOKENS   65536
#define EMBED_DIM  256
#define NUM_CODES  4096
#define COMMIT_COST 0.1f

#define MTILE   128          // tokens per workgroup in argmin kernel
#define NCHUNK  64           // codes per chunk
#define PITCH   264          // halves per LDS row (256 + 8 pad -> 528B, conflict-free b128)
#define NCHUNKS (NUM_CODES / NCHUNK)

typedef __attribute__((ext_vector_type(16))) __bf16 v16bf;
typedef __attribute__((ext_vector_type(8)))  float  v8f;

union FragBf { v16bf bf; uint4 u[2]; };

// ---- CDNA5 async global->LDS path (guarded; falls back to reg ping-pong) ----
#if defined(__has_builtin)
#  if __has_builtin(__builtin_amdgcn_global_load_async_to_lds_b128) && \
      __has_builtin(__builtin_amdgcn_s_wait_asynccnt)
#    define USE_ASYNC 1
#  endif
#endif
#ifndef USE_ASYNC
#  define USE_ASYNC 0
#endif

#if USE_ASYNC
typedef __attribute__((ext_vector_type(4))) int b128_t;
typedef __attribute__((address_space(1))) b128_t* gptr_t;   // global v4i*
typedef __attribute__((address_space(3))) b128_t* lptr_t;   // LDS v4i*
#endif

__device__ __forceinline__ unsigned short f2bf(float f) {
    union { float f; uint32_t u; } v; v.f = f;
    uint32_t u = v.u;
    uint32_t r = u + 0x7FFFu + ((u >> 16) & 1u);   // round-to-nearest-even
    return (unsigned short)(r >> 16);
}

// ---------------------------------------------------------------------------
// Kernel 1: e_sq[k] = |E_k|^2 ; E -> bf16 copy in ws ; zero counts + loss acc.
// One wave per code, 8 codes per block. grid = NUM_CODES/8.
// ---------------------------------------------------------------------------
__global__ __launch_bounds__(256)
void vq_prep(const float* __restrict__ emb, float* __restrict__ e_sq,
             unsigned short* __restrict__ emb_bf16,
             unsigned int* __restrict__ counts, float* __restrict__ sumsq) {
    const int tid  = threadIdx.x;
    const int lane = tid & 31;
    const int wave = tid >> 5;
    const int code = blockIdx.x * 8 + wave;

    const float4* p = (const float4*)(emb + (size_t)code * EMBED_DIM);
    float4 a = p[lane * 2], b = p[lane * 2 + 1];
    float s = a.x*a.x + a.y*a.y + a.z*a.z + a.w*a.w
            + b.x*b.x + b.y*b.y + b.z*b.z + b.w*b.w;
    #pragma unroll
    for (int m = 1; m < 32; m <<= 1) s += __shfl_xor(s, m, 32);
    if (lane == 0) e_sq[code] = s;

    // convert this block's 8 embedding rows (512 float4) to packed bf16
    const float4* src4 = (const float4*)emb + (size_t)blockIdx.x * 512;
    uint2*        dst2 = (uint2*)emb_bf16 + (size_t)blockIdx.x * 512;
    #pragma unroll
    for (int i = 0; i < 2; ++i) {
        int j = tid + i * 256;
        float4 v = src4[j];
        union { unsigned short h[4]; uint2 u; } pk;
        pk.h[0] = f2bf(v.x); pk.h[1] = f2bf(v.y);
        pk.h[2] = f2bf(v.z); pk.h[3] = f2bf(v.w);
        dst2[j] = pk.u;
    }

    // zero histogram + loss accumulator (fresh every launch)
    if (blockIdx.x < NUM_CODES / 256) counts[blockIdx.x * 256 + tid] = 0u;
    if (blockIdx.x == 0 && tid == 0) *sumsq = 0.0f;
}

// ---------------------------------------------------------------------------
// Kernel 2: fused bf16-WMMA distance GEMM + row argmin.
// 256 threads (8 waves, 4x2 wave grid), 128 tokens x all 4096 codes.
// B chunks double-buffered in LDS; staged from pre-converted bf16 table via
// GLOBAL_LOAD_ASYNC_TO_LDS_B128 when available (ASYNCcnt), else reg ping-pong.
// ---------------------------------------------------------------------------
__global__ __launch_bounds__(256)
void vq_argmin(const float* __restrict__ z_e,
               const unsigned short* __restrict__ emb_bf16,
               const float* __restrict__ e_sq, float* __restrict__ idx_out) {
    extern __shared__ char smem[];
    unsigned short* Alds = (unsigned short*)smem;                   // MTILE*PITCH
    unsigned short* Blds = Alds + MTILE * PITCH;                    // 2 * NCHUNK*PITCH
    float*          cmbV = (float*)(Blds + 2 * NCHUNK * PITCH);     // 2*MTILE
    int*            cmbI = (int*)(cmbV + 2 * MTILE);                // 2*MTILE

    const int tid     = threadIdx.x;
    const int lane    = tid & 31;
    const int l15     = lane & 15;
    const int halfsel = lane >> 4;
    const int wave    = tid >> 5;
    const int wave_m  = wave & 3;            // 4 waves along M (32 rows each)
    const int wave_n  = wave >> 2;           // 2 waves along N (32 cols each)
    const int row0    = blockIdx.x * MTILE;

    // per-thread slice of one B chunk: 8 x 16B. item j = tid + i*256;
    // row r = j>>5 (32 x 16B per 512B row), 16B column cw = j&31.
    // global src byte = (nbase + r)*512 + cw*16 ; LDS dst byte = r*528 + cw*16
    const int brow = tid >> 5;               // base row contribution from tid
    const int bcw  = tid & 31;

    // ---- stage A tile: z_e[row0 .. row0+127][0..255] -> bf16 LDS ----
    {
        const float4* zp = (const float4*)(z_e + (size_t)row0 * EMBED_DIM);
        #pragma unroll 4
        for (int i = 0; i < 32; ++i) {               // 8192 float4 / 256 threads
            int e4 = tid + i * 256;
            float4 v = zp[e4];
            int r = e4 >> 6;
            int c = (e4 & 63) * 4;
            union { unsigned short h[4]; uint2 u; } pk;
            pk.h[0] = f2bf(v.x); pk.h[1] = f2bf(v.y);
            pk.h[2] = f2bf(v.z); pk.h[3] = f2bf(v.w);
            *(uint2*)&Alds[r * PITCH + c] = pk.u;
        }
    }

    float minv[16];
    int   mini[16];
    #pragma unroll
    for (int s = 0; s < 16; ++s) { minv[s] = 3.4e38f; mini[s] = 0; }

    const v8f vzero = {0.f, 0.f, 0.f, 0.f, 0.f, 0.f, 0.f, 0.f};

    const char* gB = (const char*)emb_bf16;

#if USE_ASYNC
    // prologue: async-stage chunk 0 into buffer 0
    #pragma unroll
    for (int i = 0; i < 8; ++i) {
        int r = brow + i * 8;                        // rows 0..63
        const char* src = gB + (size_t)r * 512 + bcw * 16;
        char*       dst = (char*)Blds + r * (PITCH * 2) + bcw * 16;
        __builtin_amdgcn_global_load_async_to_lds_b128(
            (gptr_t)src, (lptr_t)dst, 0, 0);
    }
    __builtin_amdgcn_s_wait_asynccnt(0);
    __syncthreads();
#else
    // prologue: sync-stage chunk 0 into buffer 0
    #pragma unroll
    for (int i = 0; i < 8; ++i) {
        int r = brow + i * 8;
        uint4 d = *(const uint4*)(gB + (size_t)r * 512 + bcw * 16);
        *(uint4*)((char*)Blds + r * (PITCH * 2) + bcw * 16) = d;
    }
    __syncthreads();
#endif

    for (int ch = 0; ch < NCHUNKS; ++ch) {
        const int nbase = ch * NCHUNK;
        unsigned short* Bcur = Blds + (ch & 1) * (NCHUNK * PITCH);
        char*           Bnxt = (char*)(Blds + ((ch + 1) & 1) * (NCHUNK * PITCH));

#if USE_ASYNC
        // kick off next chunk's DMA while we compute this one
        if (ch + 1 < NCHUNKS) {
            const char* gsrc0 = gB + (size_t)(nbase + NCHUNK) * 512;
            #pragma unroll
            for (int i = 0; i < 8; ++i) {
                int r = brow + i * 8;
                __builtin_amdgcn_global_load_async_to_lds_b128(
                    (gptr_t)(gsrc0 + (size_t)r * 512 + bcw * 16),
                    (lptr_t)(Bnxt + r * (PITCH * 2) + bcw * 16), 0, 0);
            }
        }
#else
        // load next chunk into registers; stored to LDS after compute
        uint4 rb[8];
        if (ch + 1 < NCHUNKS) {
            const char* gsrc0 = gB + (size_t)(nbase + NCHUNK) * 512;
            #pragma unroll
            for (int i = 0; i < 8; ++i) {
                int r = brow + i * 8;
                rb[i] = *(const uint4*)(gsrc0 + (size_t)r * 512 + bcw * 16);
            }
        }
#endif

        v8f acc[2][2];
        acc[0][0] = vzero; acc[0][1] = vzero;
        acc[1][0] = vzero; acc[1][1] = vzero;

        // ---- 8 K-steps of 16x16x32 bf16 WMMA over D=256 ----
        #pragma unroll
        for (int dk = 0; dk < 8; ++dk) {
            const int d0 = dk * 32;
            FragBf a0, a1, b0, b1;
            {   // A frags: lane holds its row; K 0-15 or 16-31
                const int kb = halfsel ? 16 : 0;
                const unsigned short* base =
                    &Alds[(wave_m * 32 + l15) * PITCH + d0 + kb];
                a0.u[0] = *(const uint4*)(base);
                a0.u[1] = *(const uint4*)(base + 8);
                const unsigned short* base1 = base + 16 * PITCH;
                a1.u[0] = *(const uint4*)(base1);
                a1.u[1] = *(const uint4*)(base1 + 8);
            }
            {   // B frags: lane holds its column (code row); K 0-15 or 16-31
                const int kb = halfsel ? 16 : 0;
                const unsigned short* base =
                    &Bcur[(wave_n * 32 + l15) * PITCH + d0 + kb];
                b0.u[0] = *(const uint4*)(base);
                b0.u[1] = *(const uint4*)(base + 8);
                const unsigned short* base1 = base + 16 * PITCH;
                b1.u[0] = *(const uint4*)(base1);
                b1.u[1] = *(const uint4*)(base1 + 8);
            }
            acc[0][0] = __builtin_amdgcn_wmma_f32_16x16x32_bf16(
                false, a0.bf, false, b0.bf, (short)0, acc[0][0], false, false);
            acc[0][1] = __builtin_amdgcn_wmma_f32_16x16x32_bf16(
                false, a0.bf, false, b1.bf, (short)0, acc[0][1], false, false);
            acc[1][0] = __builtin_amdgcn_wmma_f32_16x16x32_bf16(
                false, a1.bf, false, b0.bf, (short)0, acc[1][0], false, false);
            acc[1][1] = __builtin_amdgcn_wmma_f32_16x16x32_bf16(
                false, a1.bf, false, b1.bf, (short)0, acc[1][1], false, false);
        }

        // ---- epilogue: score = e_sq[n] - 2*dot ; running per-row argmin ----
        #pragma unroll
        for (int nt = 0; nt < 2; ++nt) {
            const int n  = nbase + wave_n * 32 + nt * 16 + l15;
            const float es = e_sq[n];
            #pragma unroll
            for (int mt = 0; mt < 2; ++mt) {
                #pragma unroll
                for (int r = 0; r < 8; ++r) {
                    float sc = es - 2.0f * acc[mt][nt][r];
                    const int s = mt * 8 + r;
                    if (sc < minv[s]) { minv[s] = sc; mini[s] = n; }
                }
            }
        }

#if USE_ASYNC
        __builtin_amdgcn_s_wait_asynccnt(0);     // next buffer landed
        __syncthreads();                          // visible to all waves
#else
        __syncthreads();                          // everyone done with old buffer
        if (ch + 1 < NCHUNKS) {
            #pragma unroll
            for (int i = 0; i < 8; ++i) {
                int r = brow + i * 8;
                *(uint4*)(Bnxt + r * (PITCH * 2) + bcw * 16) = rb[i];
            }
        }
        __syncthreads();                          // next buffer visible
#endif
    }

    // ---- reduce across the 16 lanes sharing each row, then across wave_n ----
    #pragma unroll
    for (int s = 0; s < 16; ++s) {
        float v = minv[s]; int i = mini[s];
        #pragma unroll
        for (int m = 1; m < 16; m <<= 1) {       // stays within 16-lane half
            float ov = __shfl_xor(v, m, 32);
            int   oi = __shfl_xor(i, m, 32);
            if (ov < v || (ov == v && oi < i)) { v = ov; i = oi; }
        }
        const int mt = s >> 3, r = s & 7;
        const int rl = wave_m * 32 + mt * 16 + r + (halfsel ? 8 : 0);
        if (l15 == 0) { cmbV[wave_n * MTILE + rl] = v; cmbI[wave_n * MTILE + rl] = i; }
    }
    __syncthreads();

    if (tid < MTILE) {
        float v0 = cmbV[tid], v1 = cmbV[MTILE + tid];
        int   i0 = cmbI[tid], i1 = cmbI[MTILE + tid];
        int best = (v1 < v0 || (v1 == v0 && i1 < i0)) ? i1 : i0;
        idx_out[row0 + tid] = (float)best;       // exact for idx < 2^24
    }
}

// ---------------------------------------------------------------------------
// Kernel 3: gather z_q = E[idx] (== z_q_st numerically), commitment-loss
// partial sums, usage histogram. One wave per token.
// ---------------------------------------------------------------------------
__global__ __launch_bounds__(256)
void vq_gather(const float* __restrict__ z_e, const float* __restrict__ emb,
               const float* __restrict__ idxF, float* __restrict__ z_q,
               unsigned int* __restrict__ counts, float* __restrict__ sumsq) {
    const int tid  = threadIdx.x;
    const int lane = tid & 31;
    const int wave = tid >> 5;
    const int t    = blockIdx.x * 8 + wave;

    const int idx = (int)(idxF[t] + 0.5f);
    const float4* zp = (const float4*)(z_e + (size_t)t   * EMBED_DIM);
    const float4* ep = (const float4*)(emb + (size_t)idx * EMBED_DIM);
    float4*       op = (float4*)(z_q + (size_t)t * EMBED_DIM);

    float s = 0.f;
    #pragma unroll
    for (int i = 0; i < 2; ++i) {
        float4 a = zp[lane + i * 32];
        float4 e = ep[lane + i * 32];
        op[lane + i * 32] = e;
        float dx = e.x - a.x, dy = e.y - a.y, dz = e.z - a.z, dw = e.w - a.w;
        s += dx * dx + dy * dy + dz * dz + dw * dw;
    }
    #pragma unroll
    for (int m = 1; m < 32; m <<= 1) s += __shfl_xor(s, m, 32);
    if (lane == 0) {
        atomicAdd(sumsq, s);
        atomicAdd(&counts[idx], 1u);
    }
}

// ---------------------------------------------------------------------------
// Kernel 4: finalize commitment loss + perplexity.
// ---------------------------------------------------------------------------
__global__ __launch_bounds__(256)
void vq_final(const unsigned int* __restrict__ counts,
              const float* __restrict__ sumsq,
              float* __restrict__ out_loss, float* __restrict__ out_ppl) {
    __shared__ float red[256];
    const int tid = threadIdx.x;
    float ent = 0.f;
    for (int i = tid; i < NUM_CODES; i += 256) {
        float p = (float)counts[i] * (1.0f / (float)N_TOKENS);
        ent += p * __logf(p + 1e-10f);
    }
    red[tid] = ent;
    __syncthreads();
    for (int s = 128; s > 0; s >>= 1) {
        if (tid < s) red[tid] += red[tid + s];
        __syncthreads();
    }
    if (tid == 0) {
        *out_ppl  = __expf(-red[0]);
        *out_loss = COMMIT_COST * (*sumsq) *
                    (1.0f / ((float)N_TOKENS * (float)EMBED_DIM));
    }
}

// ---------------------------------------------------------------------------
extern "C" void kernel_launch(void* const* d_in, const int* in_sizes, int n_in,
                              void* d_out, int out_size, void* d_ws, size_t ws_size,
                              hipStream_t stream) {
    (void)in_sizes; (void)n_in; (void)out_size; (void)ws_size;
    const float* z_e = (const float*)d_in[0];
    const float* emb = (const float*)d_in[1];

    float* out  = (float*)d_out;
    float* z_q  = out;                                     // [N_TOKENS*EMBED_DIM]
    float* idxF = out + (size_t)N_TOKENS * EMBED_DIM;      // [N_TOKENS]
    float* loss = idxF + N_TOKENS;                         // [1]
    float* ppl  = loss + 1;                                // [1]

    char* ws = (char*)d_ws;
    float*          e_sq     = (float*)ws;                     // 16 KB
    unsigned int*   counts   = (unsigned int*)(ws + 16384);    // 16 KB
    float*          sumsq    = (float*)(ws + 32768);           // 4 B
    unsigned short* emb_bf16 = (unsigned short*)(ws + 49152);  // 2 MB

    const int smem = (MTILE * PITCH + 2 * NCHUNK * PITCH) * 2 + 2 * MTILE * 8;

    vq_prep  <<<NUM_CODES / 8,    256, 0,    stream>>>(emb, e_sq, emb_bf16, counts, sumsq);
    vq_argmin<<<N_TOKENS / MTILE, 256, smem, stream>>>(z_e, emb_bf16, e_sq, idxF);
    vq_gather<<<N_TOKENS / 8,     256, 0,    stream>>>(z_e, emb, idxF, z_q, counts, sumsq);
    vq_final <<<1,                256, 0,    stream>>>(counts, sumsq, loss, ppl);
}
